// MelScale_68135361184462
// MI455X (gfx1250) — matrix operand order repaired and checked by reading, not compiled, not run on hardware
//
#include <hip/hip_runtime.h>
#include <hip/hip_bf16.h>
#include <math.h>

typedef __attribute__((ext_vector_type(16))) _Float16 v16h;
typedef __attribute__((ext_vector_type(8)))  _Float16 v8h;
typedef __attribute__((ext_vector_type(4)))  _Float16 v4h;
typedef __attribute__((ext_vector_type(8)))  float    v8f;

#define WINDOW   512
#define STEP     256
#define NFREQ    256
#define NFRAMES  127
#define NSAMP    32768
#define BATCH    512
#define MTILE    32
#define SPAN     ((MTILE - 1) * STEP + WINDOW)   // 8448 samples
#define TILES_PER_BATCH 4                        // ceil(127/32)
#define ASTRIDE  520                             // padded row stride (halves)
#define EPSF     1e-8f

// ---------------------------------------------------------------------------
// Kernel 1: normalize basis rows, emit F16 copies into workspace.
// One block per frequency (256 blocks x 256 threads).
// ---------------------------------------------------------------------------
__global__ void basis_norm_kernel(const float* __restrict__ br,
                                  const float* __restrict__ bi,
                                  _Float16* __restrict__ brH,
                                  _Float16* __restrict__ biH) {
    const int f = blockIdx.x;
    const int t = threadIdx.x;
    const float* r  = br + (size_t)f * WINDOW;
    const float* im = bi + (size_t)f * WINDOW;

    float x0 = r[t], x1 = r[t + 256];
    float y0 = im[t], y1 = im[t + 256];
    float s = x0 * x0 + x1 * x1 + y0 * y0 + y1 * y1;

    #pragma unroll
    for (int off = 16; off > 0; off >>= 1) s += __shfl_down(s, off, 32);

    __shared__ float red[8];
    __shared__ float sScale;
    if ((t & 31) == 0) red[t >> 5] = s;
    __syncthreads();
    if (t == 0) {
        float tot = 0.f;
        #pragma unroll
        for (int i = 0; i < 8; ++i) tot += red[i];
        sScale = 1.f / (sqrtf(tot) + EPSF);
    }
    __syncthreads();
    const float sc = sScale;

    _Float16* dr = brH + (size_t)f * WINDOW;
    _Float16* di = biH + (size_t)f * WINDOW;
    dr[t]       = (_Float16)(x0 * sc);
    dr[t + 256] = (_Float16)(x1 * sc);
    di[t]       = (_Float16)(y0 * sc);
    di[t + 256] = (_Float16)(y1 * sc);
}

// ---------------------------------------------------------------------------
// Kernel 2: per (batch, 32-frame tile):
//   1. stage audio span to LDS (f16)
//   2. per-frame L2 norms
//   3. expand to normalized 32x512 A-matrix in LDS (padded stride)
//   4. WMMA f16 GEMM: each wave loops over two 16-freq N-tiles with 4 accumulators
//      (M-subtile x {real,imag}) -> 4 independent WMMAs per K-step.
// Grid = 512 * 4 blocks, 256 threads = 8 waves.
// ---------------------------------------------------------------------------
__global__ void __launch_bounds__(256)
stft_wmma_kernel(const float* __restrict__ audio,
                 const _Float16* __restrict__ brH,
                 const _Float16* __restrict__ biH,
                 float* __restrict__ outNorm,
                 float* __restrict__ outReal,
                 float* __restrict__ outImag) {
    __shared__ __align__(16) _Float16 sSpan[SPAN];            // 16.9 KB raw
    __shared__ __align__(16) _Float16 sA[MTILE * ASTRIDE];    // 33.3 KB normalized
    __shared__ float sInvNorm[MTILE];

    const int tid  = threadIdx.x;
    const int b    = blockIdx.x / TILES_PER_BATCH;
    const int tile = blockIdx.x % TILES_PER_BATCH;
    const int f0   = tile * MTILE;
    const int s0   = f0 * STEP;

    // ---- 1. stage raw audio span into LDS as f16 (clamp last tile) --------
    const float* arow = audio + (size_t)b * NSAMP;
    for (int i = tid; i < SPAN / 4; i += 256) {
        const int s = s0 + i * 4;
        float4 v;
        if (s + 3 < NSAMP) {
            v = *(const float4*)(arow + s);
        } else {
            v.x = arow[min(s + 0, NSAMP - 1)];
            v.y = arow[min(s + 1, NSAMP - 1)];
            v.z = arow[min(s + 2, NSAMP - 1)];
            v.w = arow[min(s + 3, NSAMP - 1)];
        }
        v4h h;
        h[0] = (_Float16)v.x; h[1] = (_Float16)v.y;
        h[2] = (_Float16)v.z; h[3] = (_Float16)v.w;
        *(v4h*)&sSpan[i * 4] = h;
    }
    __syncthreads();

    // ---- 2. per-frame L2 norms: 8 lanes per frame ---------------------------
    {
        const int g   = tid >> 3;
        const int sub = tid & 7;
        float sum = 0.f;
        const int base = g * STEP + sub * 64;
        #pragma unroll 8
        for (int k = 0; k < 64; ++k) {
            float x = (float)sSpan[base + k];
            sum += x * x;
        }
        #pragma unroll
        for (int off = 4; off > 0; off >>= 1) sum += __shfl_down(sum, off, 8);
        if (sub == 0) {
            const float nrm = sqrtf(sum);
            sInvNorm[g] = 1.f / (nrm + EPSF);
            const int f = f0 + g;
            if (f < NFRAMES) outNorm[(size_t)b * NFRAMES + f] = nrm;
        }
    }
    __syncthreads();

    // ---- 3. expand normalized A matrix: sA[row][k] = sSpan[row*256+k]*inv --
    for (int c = tid; c < (MTILE * WINDOW) / 8; c += 256) {
        const int row = c >> 6;              // 64 chunks of 8 per row
        const int kc  = (c & 63) * 8;
        const _Float16 hs = (_Float16)sInvNorm[row];
        v8h v = *(const v8h*)&sSpan[row * STEP + kc];
        #pragma unroll
        for (int e = 0; e < 8; ++e) v[e] *= hs;
        *(v8h*)&sA[row * ASTRIDE + kc] = v;
    }
    __syncthreads();

    // ---- 4. WMMA GEMM ------------------------------------------------------
    const int wave = tid >> 5;
    const int lane = tid & 31;
    const int l    = lane & 15;
    const int hi   = lane >> 4;

    for (int nt = wave; nt < 16; nt += 8) {
        const int freq = nt * 16 + l;                    // B column (freq)
        const _Float16* __restrict__ brp = brH + (size_t)freq * WINDOW;
        const _Float16* __restrict__ bip = biH + (size_t)freq * WINDOW;

        // A fragment base addresses (rows l and 16+l)
        const _Float16* a0p = &sA[(l)      * ASTRIDE + hi * 8];
        const _Float16* a1p = &sA[(16 + l) * ASTRIDE + hi * 8];

        v8f c00 = {0.f,0.f,0.f,0.f,0.f,0.f,0.f,0.f};   // msub0, real
        v8f c01 = c00;                                  // msub0, imag
        v8f c10 = c00;                                  // msub1, real
        v8f c11 = c00;                                  // msub1, imag

        #pragma unroll 4
        for (int kt = 0; kt < WINDOW / 32; ++kt) {
            const int ko = kt * 32;
            // A fragments: VGPR0-3 = K 8*hi+[0..7], VGPR4-7 = K 16+8*hi+[0..7]
            v8h a0lo = *(const v8h*)(a0p + ko);
            v8h a0hi = *(const v8h*)(a0p + ko + 16);
            v8h a1lo = *(const v8h*)(a1p + ko);
            v8h a1hi = *(const v8h*)(a1p + ko + 16);
            v16h afrag0 = __builtin_shufflevector(a0lo, a0hi,
                0,1,2,3,4,5,6,7,8,9,10,11,12,13,14,15);
            v16h afrag1 = __builtin_shufflevector(a1lo, a1hi,
                0,1,2,3,4,5,6,7,8,9,10,11,12,13,14,15);

            // B fragments: lane = column N, K = kt*32 + 8*hi + [0..7], kt*32 + 16 + 8*hi + [0..7]
            const _Float16* bpr = brp + ko + hi * 8;
            const _Float16* bpi = bip + ko + hi * 8;
            v8h br0 = *(const v8h*)(bpr);
            v8h br1 = *(const v8h*)(bpr + 16);
            v8h bi0 = *(const v8h*)(bpi);
            v8h bi1 = *(const v8h*)(bpi + 16);
            v16h bfr = __builtin_shufflevector(br0, br1,
                0,1,2,3,4,5,6,7,8,9,10,11,12,13,14,15);
            v16h bfi = __builtin_shufflevector(bi0, bi1,
                0,1,2,3,4,5,6,7,8,9,10,11,12,13,14,15);

            c00 = __builtin_amdgcn_wmma_f32_16x16x32_f16(
                      false, afrag0, false, bfr, (short)0, c00, false, false);
            c01 = __builtin_amdgcn_wmma_f32_16x16x32_f16(
                      false, afrag0, false, bfi, (short)0, c01, false, false);
            c10 = __builtin_amdgcn_wmma_f32_16x16x32_f16(
                      false, afrag1, false, bfr, (short)0, c10, false, false);
            c11 = __builtin_amdgcn_wmma_f32_16x16x32_f16(
                      false, afrag1, false, bfi, (short)0, c11, false, false);
        }

        // C/D layout: VGPR r, lane -> M = r + 8*hi, N = lane&15
        const int n = nt * 16 + l;
        #pragma unroll
        for (int r = 0; r < 8; ++r) {
            const int m0 = r + 8 * hi;
            const int fA = f0 + m0;
            if (fA < NFRAMES) {
                const size_t o = ((size_t)b * NFRAMES + fA) * NFREQ + n;
                outReal[o] = c00[r];
                outImag[o] = c01[r];
            }
            const int fB = f0 + 16 + m0;
            if (fB < NFRAMES) {
                const size_t o = ((size_t)b * NFRAMES + fB) * NFREQ + n;
                outReal[o] = c10[r];
                outImag[o] = c11[r];
            }
        }
    }
}

// ---------------------------------------------------------------------------
extern "C" void kernel_launch(void* const* d_in, const int* in_sizes, int n_in,
                              void* d_out, int out_size, void* d_ws, size_t ws_size,
                              hipStream_t stream) {
    const float* audio      = (const float*)d_in[0];
    const float* basis_real = (const float*)d_in[1];
    const float* basis_imag = (const float*)d_in[2];

    float* out = (float*)d_out;
    float* outNorm = out;                                         // 512*127
    float* outReal = out + (size_t)BATCH * NFRAMES;               // 512*127*256
    float* outImag = outReal + (size_t)BATCH * NFRAMES * NFREQ;

    _Float16* brH = (_Float16*)d_ws;                              // 256 KB
    _Float16* biH = brH + (size_t)NFREQ * WINDOW;                 // 256 KB

    basis_norm_kernel<<<NFREQ, 256, 0, stream>>>(basis_real, basis_imag, brH, biH);

    stft_wmma_kernel<<<dim3(BATCH * TILES_PER_BATCH), 256, 0, stream>>>(
        audio, brH, biH, outNorm, outReal, outImag);
}
